// MPNNLayer_7275674599958
// MI455X (gfx1250) — matrix operand order repaired and checked by reading, not compiled, not run on hardware
//
#include <hip/hip_runtime.h>

#define N_NODES 50000
#define N_EDGES 800000
#define DIM     128
#define DIM2    256
#define LN_EPS  1e-5f

// fragment-major weight buffers: layer1 = 64 frags * 512 elems, layer2 = 32 * 512
#define W1_ELEMS (64 * 512)
#define W2_ELEMS (32 * 512)
#define W_ELEMS  (W1_ELEMS + W2_ELEMS)     // 49152 bf16
#define W_BYTES  (W_ELEMS * 2)             // 98304 B
#define HROW     136                        // padded h row stride (bank-conflict free)

typedef __bf16 v8bf  __attribute__((ext_vector_type(8)));
typedef __bf16 v16bf __attribute__((ext_vector_type(16)));
typedef float  v8f   __attribute__((ext_vector_type(8)));
typedef unsigned int u32x4 __attribute__((ext_vector_type(4)));
typedef int i32x4 __attribute__((ext_vector_type(4)));
typedef int i32x8 __attribute__((ext_vector_type(8)));

#if defined(__has_builtin)
#if __has_builtin(__builtin_amdgcn_tensor_load_to_lds)
#define HAVE_TDM 1
#endif
#endif

#define WMMA_BF16(A, B, C) \
    __builtin_amdgcn_wmma_f32_16x16x32_bf16(false, (A), false, (B), (short)0, (C), false, false)

// ---------------------------------------------------------------------------
// A-fragment (16x32 bf16): lane m=lane&15 is row M; half=lane>>4 selects
// K {koff..koff+7, 16+koff..23+koff}, koff = half*8. Two 16B loads.
// ---------------------------------------------------------------------------
__device__ __forceinline__ v16bf load_a_frag(const __bf16* base, int row, int ld,
                                             int colbase, int koff) {
    const __bf16* p = base + (size_t)row * ld + colbase;
    v8bf lo = *(const v8bf*)(p + koff);
    v8bf hi = *(const v8bf*)(p + 16 + koff);
    return __builtin_shufflevector(lo, hi, 0,1,2,3,4,5,6,7,8,9,10,11,12,13,14,15);
}

// B-fragment from fragment-major LDS buffer: frag = 512 bf16, lane reads its
// contiguous 32B slice (conflict-free).
__device__ __forceinline__ v16bf load_b_lds(const __bf16* wlds, int frag, int lane) {
    const __bf16* p = wlds + frag * 512 + lane * 16;
    v8bf lo = *(const v8bf*)(p);
    v8bf hi = *(const v8bf*)(p + 8);
    return __builtin_shufflevector(lo, hi, 0,1,2,3,4,5,6,7,8,9,10,11,12,13,14,15);
}

__device__ __forceinline__ float silu(float v) {
    return v / (1.0f + __expf(-v));
}

// ---------------------------------------------------------------------------
// Weight stage-in: TDM (tensor_load_to_lds) with cooperative-copy fallback.
// ---------------------------------------------------------------------------
#ifdef HAVE_TDM
__device__ __forceinline__ void tdm_load_weights(const __bf16* gsrc, __bf16* ldst) {
    unsigned long long ga = (unsigned long long)(uintptr_t)gsrc;
    unsigned lds_off = (unsigned)(uintptr_t)ldst;   // low 32 bits = LDS offset
    const unsigned UNITS = W_BYTES / 8;             // 12288 8-byte units
    u32x4 g0 = { 1u,                                 // count=1, user mode
                 lds_off,
                 (unsigned)(ga & 0xffffffffu),
                 (unsigned)((ga >> 32) & 0x1ffffffu) | (2u << 30) };  // addr hi | type=2
    i32x8 g1 = { (int)(3u << 16),                    // workgroup_mask=0, data_size=3 (8B)
                 (int)((UNITS & 0xffffu) << 16),     // tensor_dim0[15:0]
                 (int)(((UNITS >> 16) & 0xffffu) | (1u << 16)),  // dim0 hi | tensor_dim1=1
                 (int)((UNITS & 0xffffu) << 16),     // tile_dim0
                 1,                                  // tile_dim1=1, tile_dim2=0
                 (int)UNITS,                         // tensor_dim0_stride[31:0]
                 (int)((UNITS & 0xffffu) << 16),     // stride0 hi16=0 | dim1_stride lo16
                 0 };
    i32x4 z4 = { 0, 0, 0, 0 };
#if __has_include(<hip/amd_detail/amd_gfx1250_TDM.h>)
    i32x8 z8 = { 0, 0, 0, 0, 0, 0, 0, 0 };
    __builtin_amdgcn_tensor_load_to_lds(g0, g1, z4, z4, z8, 0);
#else
    __builtin_amdgcn_tensor_load_to_lds(g0, g1, z4, z4, 0);
#endif
}
#endif

__device__ __forceinline__ void stage_weights(const __bf16* gsrc, __bf16* ldst,
                                              int wave, int tid, int nthreads) {
#ifdef HAVE_TDM
    (void)tid; (void)nthreads;
    if (wave == 0) {
        tdm_load_weights(gsrc, ldst);
        __builtin_amdgcn_s_wait_tensorcnt(0);
    }
    __syncthreads();
#else
    const v8bf* s = (const v8bf*)gsrc;
    v8bf* d = (v8bf*)ldst;
    for (int i = tid; i < W_ELEMS / 8; i += nthreads) d[i] = s[i];
    __syncthreads();
#endif
}

// ---------------------------------------------------------------------------
// Prep kernels
// ---------------------------------------------------------------------------
__global__ __launch_bounds__(256) void prep_x_kernel(const float* __restrict__ x,
                                                     __bf16* __restrict__ xbf,
                                                     float* __restrict__ aggr) {
    for (int i = blockIdx.x * blockDim.x + threadIdx.x; i < N_NODES * DIM;
         i += gridDim.x * blockDim.x) {
        xbf[i]  = (__bf16)x[i];
        aggr[i] = 0.0f;
    }
}

// Pack weights into fragment-major bf16 layout:
// frag f=(kt*8+nt), slot r=lane*16+e  ->  element W[k][n] with
// n = nt*16 + (lane&15), k = kt*32 + (e>>3)*16 + (lane>>4)*8 + (e&7).
__global__ __launch_bounds__(256) void prep_w_kernel(
    const float* __restrict__ wm1, const float* __restrict__ wm2,
    const float* __restrict__ wu1, const float* __restrict__ wu2,
    __bf16* __restrict__ wfragM, __bf16* __restrict__ wfragU) {
    int t = blockIdx.x * blockDim.x + threadIdx.x;
    int stride = gridDim.x * blockDim.x;
    for (int i = t; i < W1_ELEMS; i += stride) {
        int f = i >> 9, r = i & 511;
        int lane = r >> 4, e = r & 15;
        int kt = f >> 3, nt = f & 7;
        int n = nt * 16 + (lane & 15);
        int k = kt * 32 + ((e >> 3) << 4) + ((lane >> 4) << 3) + (e & 7);
        wfragM[i] = (__bf16)wm1[k * DIM + n];
        wfragU[i] = (__bf16)wu1[k * DIM + n];
    }
    for (int i = t; i < W2_ELEMS; i += stride) {
        int f = i >> 9, r = i & 511;
        int lane = r >> 4, e = r & 15;
        int kt = f >> 3, nt = f & 7;
        int n = nt * 16 + (lane & 15);
        int k = kt * 32 + ((e >> 3) << 4) + ((lane >> 4) << 3) + (e & 7);
        wfragM[W1_ELEMS + i] = (__bf16)wm2[k * DIM + n];
        wfragU[W1_ELEMS + i] = (__bf16)wu2[k * DIM + n];
    }
}

__global__ __launch_bounds__(256) void cvt_aggr_kernel(const float* __restrict__ aggr,
                                                       __bf16* __restrict__ aggrbf) {
    for (int i = blockIdx.x * blockDim.x + threadIdx.x; i < N_NODES * DIM;
         i += gridDim.x * blockDim.x) {
        aggrbf[i] = (__bf16)aggr[i];
    }
}

// ---------------------------------------------------------------------------
// Edge kernel: 1 wave = 32 edges (two 16-row A sub-tiles share every B frag),
// weights LDS-resident (TDM), B double-buffered to hide ds latency.
// Block = 4 waves (128 threads), 25000 tiles / 4 = 6250 blocks.
// ---------------------------------------------------------------------------
__global__ __launch_bounds__(128) void edge_msg_kernel(
    const __bf16* __restrict__ xbf, const int* __restrict__ erow,
    const int* __restrict__ ecol, const __bf16* __restrict__ wfrag,
    const float* __restrict__ bm1, const float* __restrict__ bm2,
    float* __restrict__ aggr) {
    __shared__ __bf16 wlds[W_ELEMS];        // 96KB weight fragments
    __shared__ __bf16 hsh[4][32 * HROW];    // 32-row h staging per wave

    const int wave = threadIdx.x >> 5;
    const int lane = threadIdx.x & 31;
    const int tile = blockIdx.x * 4 + wave;
    const int e0   = tile * 32;
    const int m    = lane & 15;
    const int hf   = lane >> 4;
    const int koff = hf * 8;

    stage_weights(wfrag, wlds, wave, threadIdx.x, 128);

    const int srow0 = erow[e0 + m];
    const int scol0 = ecol[e0 + m];
    const int srow1 = erow[e0 + 16 + m];
    const int scol1 = ecol[e0 + 16 + m];

    v8f acc0[8], acc1[8];
#pragma unroll
    for (int i = 0; i < 8; ++i) {
        acc0[i] = (v8f){0.f,0.f,0.f,0.f,0.f,0.f,0.f,0.f};
        acc1[i] = (v8f){0.f,0.f,0.f,0.f,0.f,0.f,0.f,0.f};
    }

    // ---- GEMM1: [32x256] x [256x128]; A prefetch 1 kt ahead, B 1 frag ahead
    v16bf a0 = load_a_frag(xbf, srow0, DIM, 0, koff);
    v16bf a1 = load_a_frag(xbf, srow1, DIM, 0, koff);
    v16bf b  = load_b_lds(wlds, 0, lane);
#pragma unroll
    for (int kt = 0; kt < 8; ++kt) {
        v16bf an0 = a0, an1 = a1;
        if (kt < 7) {
            const int kn = kt + 1;
            an0 = load_a_frag(xbf, (kn < 4) ? srow0 : scol0, DIM, (kn & 3) * 32, koff);
            an1 = load_a_frag(xbf, (kn < 4) ? srow1 : scol1, DIM, (kn & 3) * 32, koff);
        }
#pragma unroll
        for (int nt = 0; nt < 8; ++nt) {
            const int idx = kt * 8 + nt;
            v16bf bn = b;
            if (idx < 63) bn = load_b_lds(wlds, idx + 1, lane);
            acc0[nt] = WMMA_BF16(a0, b, acc0[nt]);
            acc1[nt] = WMMA_BF16(a1, b, acc1[nt]);
            b = bn;
        }
        a0 = an0; a1 = an1;
    }

    // ---- bias + SiLU -> LDS (32 x HROW bf16) ----
    __bf16* hw = hsh[wave];
#pragma unroll
    for (int nt = 0; nt < 8; ++nt) {
        const int colm = nt * 16 + m;
        const float bb = bm1[colm];
#pragma unroll
        for (int j = 0; j < 8; ++j) {
            const int row = j + hf * 8;
            hw[row * HROW + colm]        = (__bf16)silu(acc0[nt][j] + bb);
            hw[(16 + row) * HROW + colm] = (__bf16)silu(acc1[nt][j] + bb);
        }
    }

    // ---- GEMM2: [32x128] x [128x128] ----
    v8f c0[8], c1[8];
#pragma unroll
    for (int i = 0; i < 8; ++i) {
        c0[i] = (v8f){0.f,0.f,0.f,0.f,0.f,0.f,0.f,0.f};
        c1[i] = (v8f){0.f,0.f,0.f,0.f,0.f,0.f,0.f,0.f};
    }
    b = load_b_lds(wlds, 64, lane);
#pragma unroll
    for (int kt = 0; kt < 4; ++kt) {
        v16bf a20 = load_a_frag(hw, m, HROW, kt * 32, koff);
        v16bf a21 = load_a_frag(hw + 16 * HROW, m, HROW, kt * 32, koff);
#pragma unroll
        for (int nt = 0; nt < 8; ++nt) {
            const int idx = kt * 8 + nt;
            v16bf bn = b;
            if (idx < 31) bn = load_b_lds(wlds, 64 + idx + 1, lane);
            c0[nt] = WMMA_BF16(a20, b, c0[nt]);
            c1[nt] = WMMA_BF16(a21, b, c1[nt]);
            b = bn;
        }
    }

    // ---- +bm2, atomic scatter into aggr[row[e]] ----
    float b2[8];
#pragma unroll
    for (int nt = 0; nt < 8; ++nt) b2[nt] = bm2[nt * 16 + m];
#pragma unroll
    for (int j = 0; j < 8; ++j) {
        const int dst0 = erow[e0 + j + hf * 8];
        const int dst1 = erow[e0 + 16 + j + hf * 8];
        float* ar0 = aggr + (size_t)dst0 * DIM;
        float* ar1 = aggr + (size_t)dst1 * DIM;
#pragma unroll
        for (int nt = 0; nt < 8; ++nt) {
            unsafeAtomicAdd(ar0 + nt * 16 + m, c0[nt][j] + b2[nt]);
            unsafeAtomicAdd(ar1 + nt * 16 + m, c1[nt][j] + b2[nt]);
        }
    }
}

// ---------------------------------------------------------------------------
// Node kernel: 1 wave = 16 nodes, upd_in = [x||aggr]; B double-buffered.
// ---------------------------------------------------------------------------
__global__ __launch_bounds__(256) void node_upd_kernel(
    const __bf16* __restrict__ xbf, const __bf16* __restrict__ aggrbf,
    const float* __restrict__ x, const __bf16* __restrict__ wfrag,
    const float* __restrict__ bu1, const float* __restrict__ bu2,
    float* __restrict__ yout) {
    __shared__ __bf16 wlds[W_ELEMS];
    __shared__ __bf16 hsh[8][16 * HROW];

    const int wave = threadIdx.x >> 5;
    const int lane = threadIdx.x & 31;
    const int tile = blockIdx.x * 8 + wave;
    const int m    = lane & 15;
    const int hf   = lane >> 4;
    const int koff = hf * 8;

    stage_weights(wfrag, wlds, wave, threadIdx.x, 256);
    if (tile >= N_NODES / 16) return;  // wave-uniform exit (after barrier)

    const int n0   = tile * 16;
    const int node = n0 + m;

    v8f acc[8];
#pragma unroll
    for (int i = 0; i < 8; ++i) acc[i] = (v8f){0.f,0.f,0.f,0.f,0.f,0.f,0.f,0.f};

    v16bf a = load_a_frag(xbf, node, DIM, 0, koff);
    v16bf b = load_b_lds(wlds, 0, lane);
#pragma unroll
    for (int kt = 0; kt < 8; ++kt) {
        v16bf an = a;
        if (kt < 7) {
            const int kn = kt + 1;
            const __bf16* src = (kn < 4) ? xbf : aggrbf;
            an = load_a_frag(src, node, DIM, (kn & 3) * 32, koff);
        }
#pragma unroll
        for (int nt = 0; nt < 8; ++nt) {
            const int idx = kt * 8 + nt;
            v16bf bn = b;
            if (idx < 63) bn = load_b_lds(wlds, idx + 1, lane);
            acc[nt] = WMMA_BF16(a, b, acc[nt]);
            b = bn;
        }
        a = an;
    }

    __bf16* hw = hsh[wave];
#pragma unroll
    for (int nt = 0; nt < 8; ++nt) {
        const int colm = nt * 16 + m;
        const float bb = bu1[colm];
#pragma unroll
        for (int j = 0; j < 8; ++j) {
            const int row = j + hf * 8;
            hw[row * HROW + colm] = (__bf16)silu(acc[nt][j] + bb);
        }
    }

    v8f acc2[8];
#pragma unroll
    for (int i = 0; i < 8; ++i) acc2[i] = (v8f){0.f,0.f,0.f,0.f,0.f,0.f,0.f,0.f};
    b = load_b_lds(wlds, 64, lane);
#pragma unroll
    for (int kt = 0; kt < 4; ++kt) {
        v16bf a2 = load_a_frag(hw, m, HROW, kt * 32, koff);
#pragma unroll
        for (int nt = 0; nt < 8; ++nt) {
            const int idx = kt * 8 + nt;
            v16bf bn = b;
            if (idx < 31) bn = load_b_lds(wlds, 64 + idx + 1, lane);
            acc2[nt] = WMMA_BF16(a2, b, acc2[nt]);
            b = bn;
        }
    }

    float b2[8];
#pragma unroll
    for (int nt = 0; nt < 8; ++nt) b2[nt] = bu2[nt * 16 + m];
#pragma unroll
    for (int j = 0; j < 8; ++j) {
        const int nrow = n0 + j + hf * 8;
        const float* xr = x + (size_t)nrow * DIM;
        float* yr = yout + (size_t)nrow * DIM;
#pragma unroll
        for (int nt = 0; nt < 8; ++nt) {
            const int colm = nt * 16 + m;
            yr[colm] = acc2[nt][j] + b2[nt] + xr[colm];
        }
    }
}

// ---------------------------------------------------------------------------
// LayerNorm, in-place: one wave per row, float4 per lane.
// ---------------------------------------------------------------------------
__global__ __launch_bounds__(256) void ln_kernel(float* __restrict__ y,
                                                 const float* __restrict__ gamma,
                                                 const float* __restrict__ beta) {
    const int row  = blockIdx.x * 8 + (threadIdx.x >> 5);
    const int lane = threadIdx.x & 31;
    if (row >= N_NODES) return;
    float4 v = *(const float4*)(y + (size_t)row * DIM + lane * 4);
    float s  = v.x + v.y + v.z + v.w;
    float ss = v.x * v.x + v.y * v.y + v.z * v.z + v.w * v.w;
#pragma unroll
    for (int mk = 16; mk >= 1; mk >>= 1) {
        s  += __shfl_xor(s, mk, 32);
        ss += __shfl_xor(ss, mk, 32);
    }
    const float mu  = s * (1.0f / 128.0f);
    const float var = ss * (1.0f / 128.0f) - mu * mu;
    const float inv = rsqrtf(var + LN_EPS);
    float4 g = *(const float4*)(gamma + lane * 4);
    float4 b = *(const float4*)(beta + lane * 4);
    v.x = (v.x - mu) * inv * g.x + b.x;
    v.y = (v.y - mu) * inv * g.y + b.y;
    v.z = (v.z - mu) * inv * g.z + b.z;
    v.w = (v.w - mu) * inv * g.w + b.w;
    *(float4*)(y + (size_t)row * DIM + lane * 4) = v;
}

// ---------------------------------------------------------------------------
extern "C" void kernel_launch(void* const* d_in, const int* in_sizes, int n_in,
                              void* d_out, int out_size, void* d_ws, size_t ws_size,
                              hipStream_t stream) {
    (void)in_sizes; (void)n_in; (void)out_size; (void)ws_size;
    const float* x    = (const float*)d_in[0];
    const int*   eidx = (const int*)d_in[1];
    const float* wm1  = (const float*)d_in[2];
    const float* bm1  = (const float*)d_in[3];
    const float* wm2  = (const float*)d_in[4];
    const float* bm2  = (const float*)d_in[5];
    const float* wu1  = (const float*)d_in[6];
    const float* bu1  = (const float*)d_in[7];
    const float* wu2  = (const float*)d_in[8];
    const float* bu2  = (const float*)d_in[9];
    const float* gmm  = (const float*)d_in[10];
    const float* bta  = (const float*)d_in[11];
    float* out = (float*)d_out;

    char* ws = (char*)d_ws;
    size_t off = 0;
    auto take = [&](size_t bytes) -> void* {
        void* p = ws + off;
        off += (bytes + 255) & ~(size_t)255;
        return p;
    };
    const size_t ND = (size_t)N_NODES * DIM;
    __bf16* xbf    = (__bf16*)take(ND * 2);
    __bf16* aggrbf = (__bf16*)take(ND * 2);
    float*  aggr   = (float*)take(ND * 4);
    __bf16* wfragM = (__bf16*)take((size_t)W_ELEMS * 2);
    __bf16* wfragU = (__bf16*)take((size_t)W_ELEMS * 2);

    prep_x_kernel<<<4096, 256, 0, stream>>>(x, xbf, aggr);
    prep_w_kernel<<<192, 256, 0, stream>>>(wm1, wm2, wu1, wu2, wfragM, wfragU);
    edge_msg_kernel<<<N_EDGES / 32 / 4, 128, 0, stream>>>(
        xbf, eidx, eidx + N_EDGES, wfragM, bm1, bm2, aggr);
    cvt_aggr_kernel<<<4096, 256, 0, stream>>>(aggr, aggrbf);
    node_upd_kernel<<<(N_NODES / 16 + 7) / 8, 256, 0, stream>>>(
        xbf, aggrbf, x, wfragU, bu1, bu2, out);
    ln_kernel<<<(N_NODES + 7) / 8, 256, 0, stream>>>(out, gmm, bta);
}
